// BatchMixingLoss_25821343383695
// MI455X (gfx1250) — matrix operand order repaired and checked by reading, not compiled, not run on hardware
//
#include <hip/hip_runtime.h>
#include <hip/hip_bf16.h>

#define N_CELLS 8192
#define DIM     128
#define KNN     15

typedef __attribute__((ext_vector_type(16))) __bf16 v16bf;
typedef __attribute__((ext_vector_type(8)))  __bf16 v8bf;
typedef __attribute__((ext_vector_type(8)))  float  v8f;

#if defined(__HIP_DEVICE_COMPILE__) && !__has_builtin(__builtin_amdgcn_wmma_f32_16x16x32_bf16)
#error "missing wmma bf16 builtin on this toolchain"
#endif

#if defined(__HIP_DEVICE_COMPILE__)
__device__ __forceinline__ v8f wmma_bf16(v16bf a, v16bf b, v8f c) {
  // (neg_a, A, neg_b, B, c_mod, C, reuse_a, reuse_b)
  return __builtin_amdgcn_wmma_f32_16x16x32_bf16(false, a, false, b, (short)0, c,
                                                 false, false);
}
#else
__device__ v8f wmma_bf16(v16bf a, v16bf b, v8f c);  // host stub, never emitted
#endif

__device__ __forceinline__ v16bf cat8(v8bf a, v8bf b) {
  return __builtin_shufflevector(a, b, 0, 1, 2, 3, 4, 5, 6, 7,
                                 8, 9, 10, 11, 12, 13, 14, 15);
}

// ------------- kernel 1: fp32 row sqnorms (+ optional bf16 hi/lo split) ------
template <bool CONV>
__global__ __launch_bounds__(256)
void bml_prep(const float* __restrict__ emb, float* __restrict__ sqn,
              __bf16* __restrict__ hi, __bf16* __restrict__ lo) {
  int r = blockIdx.x * blockDim.x + threadIdx.x;
  if (r >= N_CELLS) return;
  const float4* p = reinterpret_cast<const float4*>(emb + (size_t)r * DIM);
  __bf16* hp = CONV ? hi + (size_t)r * DIM : nullptr;
  __bf16* lp = CONV ? lo + (size_t)r * DIM : nullptr;
  float s = 0.f;
#pragma unroll
  for (int i = 0; i < DIM / 4; ++i) {
    float4 v = p[i];
    s += v.x * v.x + v.y * v.y + v.z * v.z + v.w * v.w;
    if (CONV) {
      float xs[4] = {v.x, v.y, v.z, v.w};
#pragma unroll
      for (int e = 0; e < 4; ++e) {
        __bf16 h = (__bf16)xs[e];
        hp[4 * i + e] = h;
        lp[4 * i + e] = (__bf16)(xs[e] - (float)h);
      }
    }
  }
  sqn[r] = s;
}

// ------------- kernel 2: fused distance GEMM + top-15 + entropy --------------
// One wave (32 lanes) per 32-row block. Lane L owns local row L for the top-k
// scan. A tiles (32x128, bf16 hi/lo split) resident in VGPRs for all 512 tiles.
template <bool PRE>
__global__ __launch_bounds__(32)
void bml_main(const float* __restrict__ emb,
              const __bf16* __restrict__ ehi, const __bf16* __restrict__ elo,
              const int* __restrict__ labels, const float* __restrict__ sqn,
              float* __restrict__ ent_out) {
  __shared__ float tile[32][20];  // stride 20 dwords: 16B-aligned rows

  const int t      = threadIdx.x;  // 0..31
  const int i0     = blockIdx.x * 32;
  const int lanelo = t & 15;
  const int khalf  = t >> 4;

  // ---- resident A operands: rows [i0, i0+32) ----
  // A 16x32 bf16, k order used here (B matches): lane m = row m; v0..3 = K[16*khalf .. +8),
  // v4..7 = K[16*khalf+8 .. +8)   (per K-chunk of 32).
  v16bf ah[2][4], al[2][4];
#pragma unroll
  for (int I = 0; I < 2; ++I) {
    const size_t rowoff = (size_t)(i0 + 16 * I + lanelo) * DIM;
#pragma unroll
    for (int c = 0; c < 4; ++c) {
      const int s1 = 32 * c + 16 * khalf;
      const int s2 = s1 + 8;
      if (PRE) {
        v8bf h1 = *reinterpret_cast<const v8bf*>(ehi + rowoff + s1);
        v8bf h2 = *reinterpret_cast<const v8bf*>(ehi + rowoff + s2);
        v8bf l1 = *reinterpret_cast<const v8bf*>(elo + rowoff + s1);
        v8bf l2 = *reinterpret_cast<const v8bf*>(elo + rowoff + s2);
        ah[I][c] = cat8(h1, h2);
        al[I][c] = cat8(l1, l2);
      } else {
        const float* ar = emb + rowoff;
#pragma unroll
        for (int e = 0; e < 8; ++e) {
          float x1 = ar[s1 + e], x2 = ar[s2 + e];
          __bf16 h1 = (__bf16)x1, h2 = (__bf16)x2;
          ah[I][c][e]     = h1;  al[I][c][e]     = (__bf16)(x1 - (float)h1);
          ah[I][c][8 + e] = h2;  al[I][c][8 + e] = (__bf16)(x2 - (float)h2);
        }
      }
    }
  }

  // per-lane row sqnorms for the C-tile rows this lane's VGPRs hold
  float rsq[2][8];
#pragma unroll
  for (int I = 0; I < 2; ++I)
#pragma unroll
    for (int v = 0; v < 8; ++v)
      rsq[I][v] = sqn[i0 + 16 * I + v + 8 * khalf];

  // ---- top-15 state (unsorted list + running max), register resident ----
  float bd[KNN];
  int   bj[KNN];
#pragma unroll
  for (int i = 0; i < KNN; ++i) { bd[i] = 3.0e38f; bj[i] = 0; }
  float cmax = 3.0e38f;
  int   cpos = 0;

  // ---- stream over all 512 column tiles ----
  for (int j0 = 0; j0 < N_CELLS; j0 += 16) {
    const size_t boff = (size_t)(j0 + lanelo) * DIM + 16 * khalf;
    float sj = sqn[j0 + lanelo];

    v8f acc0 = {};
    v8f acc1 = {};

    if (PRE) {
      const __bf16* bph = ehi + boff;
      const __bf16* bpl = elo + boff;
      if (j0 + 16 < N_CELLS) {  // pull next tile's B panel toward the WGP
        __builtin_prefetch(bph + (size_t)16 * DIM, 0, 3);
        __builtin_prefetch(bpl + (size_t)16 * DIM, 0, 3);
      }
      // 2-deep software pipeline: chunk c+1 loads issue before chunk c WMMAs
      v16bf bh_cur = cat8(*reinterpret_cast<const v8bf*>(bph),
                          *reinterpret_cast<const v8bf*>(bph + 8));
      v16bf bl_cur = cat8(*reinterpret_cast<const v8bf*>(bpl),
                          *reinterpret_cast<const v8bf*>(bpl + 8));
#pragma unroll
      for (int c = 0; c < 4; ++c) {
        v16bf bh_nxt, bl_nxt;
        if (c < 3) {
          const __bf16* nh = bph + 32 * (c + 1);
          const __bf16* nl = bpl + 32 * (c + 1);
          bh_nxt = cat8(*reinterpret_cast<const v8bf*>(nh),
                        *reinterpret_cast<const v8bf*>(nh + 8));
          bl_nxt = cat8(*reinterpret_cast<const v8bf*>(nl),
                        *reinterpret_cast<const v8bf*>(nl + 8));
        }
        acc0 = wmma_bf16(ah[0][c], bh_cur, acc0);
        acc0 = wmma_bf16(ah[0][c], bl_cur, acc0);
        acc0 = wmma_bf16(al[0][c], bh_cur, acc0);
        acc1 = wmma_bf16(ah[1][c], bh_cur, acc1);
        acc1 = wmma_bf16(ah[1][c], bl_cur, acc1);
        acc1 = wmma_bf16(al[1][c], bh_cur, acc1);
        if (c < 3) { bh_cur = bh_nxt; bl_cur = bl_nxt; }
      }
    } else {
      const float* bp = emb + boff;
#pragma unroll
      for (int c = 0; c < 4; ++c) {
        v16bf bh, bl;
#pragma unroll
        for (int e = 0; e < 16; ++e) {
          float x = bp[32 * c + e];
          __bf16 h = (__bf16)x;
          bh[e] = h;
          bl[e] = (__bf16)(x - (float)h);
        }
        acc0 = wmma_bf16(ah[0][c], bh, acc0);
        acc0 = wmma_bf16(ah[0][c], bl, acc0);
        acc0 = wmma_bf16(al[0][c], bh, acc0);
        acc1 = wmma_bf16(ah[1][c], bh, acc1);
        acc1 = wmma_bf16(ah[1][c], bl, acc1);
        acc1 = wmma_bf16(al[1][c], bh, acc1);
      }
    }

    // distances -> LDS (transpose C layout to row-per-lane)
    const int j = j0 + lanelo;
#pragma unroll
    for (int v = 0; v < 8; ++v) {
      const int m0 = v + 8 * khalf;
      float d0 = rsq[0][v] + sj - 2.0f * acc0[v];
      if (i0 + m0 == j) d0 += 1.0e10f;  // self distance
      tile[m0][lanelo] = d0;
      float d1 = rsq[1][v] + sj - 2.0f * acc1[v];
      if (i0 + 16 + m0 == j) d1 += 1.0e10f;
      tile[16 + m0][lanelo] = d1;
    }
    __syncthreads();

    // lane t scans its own row's 16 new distances
#pragma unroll
    for (int g = 0; g < 4; ++g) {
      float4 q = *reinterpret_cast<const float4*>(&tile[t][4 * g]);
      float dv[4] = {q.x, q.y, q.z, q.w};
#pragma unroll
      for (int e = 0; e < 4; ++e) {
        float d = dv[e];
        if (d < cmax) {  // rare branch: ~110 inserts / 8192 elements
          int   jj = j0 + 4 * g + e;
          float nm = -1.0f;
          int   np = 0;
#pragma unroll
          for (int i = 0; i < KNN; ++i) {
            if (i == cpos) { bd[i] = d; bj[i] = jj; }
            if (bd[i] > nm) { nm = bd[i]; np = i; }
          }
          cmax = nm;
          cpos = np;
        }
      }
    }
    __syncthreads();
  }

  // ---- epilogue: softmax over the 15 kept logits, 3-bin entropy ----
  float dmin = bd[0];
#pragma unroll
  for (int i = 1; i < KNN; ++i) dmin = fminf(dmin, bd[i]);

  float S = 0.f, p0 = 0.f, p1 = 0.f, p2 = 0.f;
#pragma unroll
  for (int i = 0; i < KNN; ++i) {
    float w = __expf(dmin - bd[i]);
    S += w;
    int lbl = labels[bj[i]];
    p0 += (lbl == 0) ? w : 0.f;
    p1 += (lbl == 1) ? w : 0.f;
    p2 += (lbl == 2) ? w : 0.f;
  }
  // masked softmax mass ~= full mass -> EPS*Z term reduces to relative 1e-8
  float inv = 1.0f / (S * (1.0f + 1e-8f));
  p0 *= inv; p1 *= inv; p2 *= inv;
  float ent = -(p0 * __logf(p0 + 1e-8f) + p1 * __logf(p1 + 1e-8f) +
                p2 * __logf(p2 + 1e-8f));
  ent_out[i0 + t] = ent * (1.0f / (logf(3.0f) + 1e-8f));
}

// ---------------- kernel 3: deterministic reduction ----------------
__global__ __launch_bounds__(256)
void bml_reduce(const float* __restrict__ ent, float* __restrict__ out) {
  __shared__ float sm[256];
  float s = 0.f;
  for (int i = threadIdx.x; i < N_CELLS; i += 256) s += ent[i];
  sm[threadIdx.x] = s;
  __syncthreads();
  for (int off = 128; off > 0; off >>= 1) {
    if (threadIdx.x < (unsigned)off) sm[threadIdx.x] += sm[threadIdx.x + off];
    __syncthreads();
  }
  if (threadIdx.x == 0) out[0] = -(sm[0] / (float)N_CELLS);
}

extern "C" void kernel_launch(void* const* d_in, const int* in_sizes, int n_in,
                              void* d_out, int out_size, void* d_ws, size_t ws_size,
                              hipStream_t stream) {
  (void)in_sizes; (void)n_in; (void)out_size;
  const float* emb    = (const float*)d_in[0];
  const int*   labels = (const int*)d_in[1];
  float*       out    = (float*)d_out;

  float*  sqn = (float*)d_ws;                       // 8192 f32
  float*  ent = sqn + N_CELLS;                      // 8192 f32
  __bf16* hi  = (__bf16*)(ent + N_CELLS);           // 8192*128 bf16
  __bf16* lo  = hi + (size_t)N_CELLS * DIM;         // 8192*128 bf16

  const size_t need_pre =
      2 * (size_t)N_CELLS * sizeof(float) + 2 * (size_t)N_CELLS * DIM * sizeof(__bf16);
  const bool pre = (ws_size >= need_pre);

  if (pre) {
    bml_prep<true><<<dim3(N_CELLS / 256), dim3(256), 0, stream>>>(emb, sqn, hi, lo);
    bml_main<true><<<dim3(N_CELLS / 32), dim3(32), 0, stream>>>(emb, hi, lo, labels,
                                                                sqn, ent);
  } else {
    bml_prep<false><<<dim3(N_CELLS / 256), dim3(256), 0, stream>>>(emb, sqn, nullptr,
                                                                   nullptr);
    bml_main<false><<<dim3(N_CELLS / 32), dim3(32), 0, stream>>>(emb, nullptr, nullptr,
                                                                 labels, sqn, ent);
  }
  bml_reduce<<<dim3(1), dim3(256), 0, stream>>>(ent, out);
}
